// modConv_50465865728276
// MI455X (gfx1250) — compile-verified, matrix operand order checked by reading, not run on hardware
//
#include <hip/hip_runtime.h>

typedef __attribute__((ext_vector_type(16))) __bf16 v16bf;
typedef __attribute__((ext_vector_type(8)))  float  v8f;
typedef __attribute__((ext_vector_type(4)))  unsigned int u32x4;
typedef __attribute__((ext_vector_type(8)))  int    i32x8;
typedef __attribute__((ext_vector_type(4)))  int    i32x4;

union BfOp {
  uint4 q[2];
  v16bf v;
};

#define B_      8
#define FI_     512
#define FO_     512
#define HW_     64
#define CHUNK   64
#define XPITCH  72          // CHUNK + 8 bf16 pad -> kills LDS bank conflicts
#define YP      132         // padded y: 129 valid + 1 front / 2 back halo

// workspace layout (bytes)
#define WS_S_OFF   0u
#define WS_A_OFF   16384ull                       // 8*512*4
#define WS_Y_OFF   37765120ull                    // A: 8*512*9*512*2 = 37748736

__device__ __forceinline__ v8f wmma_bf16(const BfOp& a, const BfOp& b, v8f c) {
  // 8 args: (neg_a, A, neg_b, B, c_mod, C, reuse_a, reuse_b)
  return __builtin_amdgcn_wmma_f32_16x16x32_bf16(false, a.v, false, b.v,
                                                 (short)0, c, false, false);
}

// ---------------------------------------------------------------- kernel 1
__global__ __launch_bounds__(256)
void style_kernel(const float* __restrict__ style, const float* __restrict__ mw,
                  const float* __restrict__ mb, float* __restrict__ s) {
  int t = blockIdx.x * 256 + threadIdx.x;          // 4096 = B*FI
  int b = t >> 9, i = t & 511;
  const float c = 0.04419417382415922f;            // 512^-0.5
  float acc = mb[i];
  for (int d = 0; d < 512; ++d)
    acc += style[b * 512 + d] * mw[i * 512 + d] * c;
  s[t] = acc;
}

// ---------------------------------------------------------------- kernel 2
// per (b,fo): demodulate and pack bf16 weights A[b][fo][tap:9][fi]
// tap order t0..t8 -> w[ky][kx]: (2,2)(2,0)(2,1)(0,2)(1,2)(0,0)(1,0)(0,1)(1,1)
__global__ __launch_bounds__(256)
void modw_kernel(const float* __restrict__ w, const float* __restrict__ s,
                 unsigned short* __restrict__ Aw) {
  __shared__ float red[256];
  __shared__ float dsh;
  const int b  = blockIdx.x >> 9;
  const int fo = blockIdx.x & 511;
  const int tid = threadIdx.x;
  const float wscale = 0.014731391274719739f;      // (512*9)^-0.5

  float acc = 0.f;
  for (int idx = tid; idx < FI_ * 9; idx += 256) {
    int fi = idx / 9, k9 = idx % 9;
    float val = w[((size_t)fo * FI_ + fi) * 9 + k9] * (wscale * s[b * FI_ + fi]);
    acc += val * val;
  }
  red[tid] = acc;
  __syncthreads();
  for (int off = 128; off > 0; off >>= 1) {
    if (tid < off) red[tid] += red[tid + off];
    __syncthreads();
  }
  if (tid == 0) dsh = rsqrtf(red[0] + 1e-8f);
  __syncthreads();
  const float dm = dsh;

  const int tky[9] = {2, 2, 2, 0, 1, 0, 1, 0, 1};
  const int tkx[9] = {2, 0, 1, 2, 2, 0, 0, 1, 1};
  __bf16* Ab = (__bf16*)Aw;
  for (int idx = tid; idx < 9 * FI_; idx += 256) {
    int t = idx >> 9, fi = idx & 511;
    float val = w[(((size_t)fo * FI_ + fi) * 3 + tky[t]) * 3 + tkx[t]] *
                (wscale * s[b * FI_ + fi]) * dm;
    Ab[(((size_t)b * FO_ + fo) * 9 + t) * FI_ + fi] = (__bf16)val;
  }
}

// ---------------------------------------------------------------- kernel 3
__global__ void zero_border_kernel(float* __restrict__ ypad) {
  float* yb = ypad + (size_t)blockIdx.x * (YP * YP);
  for (int i = threadIdx.x; i < 792; i += 256) {
    int r, c;
    if (i < 396) { int k = i / YP; r = (k == 0) ? 0 : (129 + k); c = i % YP; }
    else { int j = i - 396; int k = j / YP; c = (k == 0) ? 0 : (129 + k); r = j % YP; }
    yb[r * YP + c] = 0.0f;
  }
}

// ---------------------------------------------------------------- kernel 4
// implicit-GEMM phase-decomposed stride-2 tconv.
// Block: 8 waves = 4 u-rows x 2 fo16 blocks.  Per fi-chunk:
//   - wave0 issues one TDM tensor_load_to_lds for the A tile
//     (288 rows x 128B, LDS-padded +16B per row == XPITCH),
//   - all threads vector-stage the x tile (f32 -> bf16),
//   - 2 k-steps x (26 ds_load_b128 + 9 v_wmma_f32_16x16x32_bf16).
__global__ __launch_bounds__(256)
void conv_kernel(const float* __restrict__ x, const unsigned short* __restrict__ Aw,
                 float* __restrict__ ypad) {
  __shared__ __align__(16) __bf16 xs[5 * 17 * XPITCH];
  __shared__ __align__(16) __bf16 as[32 * 9 * XPITCH];

  const int b   = blockIdx.z;
  const int fo0 = blockIdx.y * 32;
  const int u0  = (blockIdx.x / 5) * 4;            // 0..64
  const int v0  = (blockIdx.x % 5) * 16;           // 0..64

  const int tid = threadIdx.x;
  const int wave = tid >> 5, lane = tid & 31;
  const int uw = wave & 3, fb = wave >> 2;
  const int laneN = lane & 15, laneH = lane >> 4;

  v8f c00 = {}, c10 = {}, c01 = {}, c11 = {};
  const char* Abase = (const char*)Aw;
  // flat shared pointer: low 32 bits hold the LDS byte offset
  const unsigned ldsA = (unsigned)(size_t)(void*)as;

  for (int fc = 0; fc < FI_; fc += CHUNK) {
    // ---- TDM: DMA the A chunk (fo 32 x tap 9 = 288 rows of 128B) to LDS.
    // pad_interval=4 (pad after 32 DWORDs), pad_amount=3 (+4 DWORDs) => 144B
    // LDS pitch, matching XPITCH.  Tracked by TENSORcnt.
    if (wave == 0) {
      unsigned long long ga = (unsigned long long)(size_t)Abase +
          (((((unsigned long long)b * FO_ + fo0) * 9) * FI_ + fc) * 2ull);
      u32x4 g0;
      g0[0] = 1u;                                       // count=1 (valid D#)
      g0[1] = ldsA;                                     // lds_addr (bytes)
      g0[2] = (unsigned)(ga & 0xffffffffull);           // global_addr lo
      g0[3] = (unsigned)((ga >> 32) & 0x01ffffffull)    // global_addr hi
              | (2u << 30);                             // type=2 ("image")
      i32x8 g1;
      g1[0] = (1 << 16)      // data_size = 2 bytes
            | (1 << 20)      // pad_enable
            | (4 << 22)      // pad_interval: 8<<4 = 128B before pad
            | (3 << 25);     // pad_amount: 4 DWORDs = 16B
      g1[1] = (int)((512u & 0xffffu) << 16);            // tensor_dim0 lo16
      g1[2] = (int)((512u >> 16) | ((4608u & 0xffffu) << 16)); // d0 hi | d1 lo
      g1[3] = (int)((4608u >> 16) | (64u << 16));       // d1 hi | tile_dim0=64
      g1[4] = (int)288;                                 // tile_dim1=288, tile_dim2=0
      g1[5] = (int)512;                                 // tensor_dim0_stride lo
      g1[6] = 0;                                        // stride0 hi | stride1 lo
      g1[7] = 0;
      i32x4 z4 = {0, 0, 0, 0};
#if __clang_major__ >= 23
      i32x8 z8 = {0, 0, 0, 0, 0, 0, 0, 0};
      __builtin_amdgcn_tensor_load_to_lds(g0, g1, z4, z4, z8, 0);
#else
      __builtin_amdgcn_tensor_load_to_lds(g0, g1, z4, z4, 0);
#endif
    }

    // ---- vector path: stage x tile (5 rows x 17 cols x CHUNK fi) as bf16
    for (int idx = tid; idx < 5 * 17 * CHUNK; idx += 256) {
      int fiL = idx & (CHUNK - 1);
      int rc  = idx >> 6;
      int r = rc / 17, c = rc % 17;
      int gr = u0 - 1 + r, gc = v0 - 1 + c;
      float val = 0.0f;
      if ((unsigned)gr < 64u && (unsigned)gc < 64u)
        val = x[(((size_t)b * FI_ + fc + fiL) * 64 + gr) * 64 + gc];
      xs[rc * XPITCH + fiL] = (__bf16)val;
    }
    if (fc + CHUNK < FI_)
      __builtin_prefetch(x + ((size_t)b * FI_ + fc + CHUNK) * 4096 + tid * 16, 0, 1);

    if (wave == 0) __builtin_amdgcn_s_wait_tensorcnt(0);
    __syncthreads();

    #pragma unroll
    for (int ks = 0; ks < 2; ++ks) {
      const int fi0 = ks * 32 + laneH * 8;
      BfOp Bop[2][2];
      #pragma unroll
      for (int a = 0; a < 2; ++a)
        #pragma unroll
        for (int d = 0; d < 2; ++d) {
          const __bf16* p = xs + ((uw + a) * 17 + (laneN + d)) * XPITCH + fi0;
          Bop[a][d].q[0] = *(const uint4*)p;
          Bop[a][d].q[1] = *(const uint4*)(p + 16);
        }
      BfOp Ao[9];
      #pragma unroll
      for (int t = 0; t < 9; ++t) {
        const __bf16* p = as + ((fb * 16 + laneN) * 9 + t) * XPITCH + fi0;
        Ao[t].q[0] = *(const uint4*)p;
        Ao[t].q[1] = *(const uint4*)(p + 16);
      }
      c00 = wmma_bf16(Ao[0], Bop[0][0], c00);
      c00 = wmma_bf16(Ao[1], Bop[0][1], c00);
      c01 = wmma_bf16(Ao[2], Bop[0][1], c01);
      c00 = wmma_bf16(Ao[3], Bop[1][0], c00);
      c10 = wmma_bf16(Ao[4], Bop[1][0], c10);
      c00 = wmma_bf16(Ao[5], Bop[1][1], c00);
      c10 = wmma_bf16(Ao[6], Bop[1][1], c10);
      c01 = wmma_bf16(Ao[7], Bop[1][1], c01);
      c11 = wmma_bf16(Ao[8], Bop[1][1], c11);
    }
    __syncthreads();
  }

  // phase stores into padded y (pad offset +1 in each dim)
  const int u = u0 + uw;
  const int v = v0 + laneN;
  const int ch0 = fo0 + fb * 16 + laneH * 8;
  #pragma unroll
  for (int e = 0; e < 8; ++e) {
    float* yb = ypad + ((size_t)b * FO_ + ch0 + e) * (YP * YP);
    if (u <= 64 && v <= 64) yb[(size_t)(2 * u + 1) * YP + (2 * v + 1)] = c00[e];
    if (u <= 63 && v <= 64) yb[(size_t)(2 * u + 2) * YP + (2 * v + 1)] = c10[e];
    if (u <= 64 && v <= 63) yb[(size_t)(2 * u + 1) * YP + (2 * v + 2)] = c01[e];
    if (u <= 63 && v <= 63) yb[(size_t)(2 * u + 2) * YP + (2 * v + 2)] = c11[e];
  }
}

// ---------------------------------------------------------------- kernel 5
__global__ __launch_bounds__(256)
void fir_kernel(const float* __restrict__ ypad, float* __restrict__ out) {
  size_t q = (size_t)blockIdx.x * 256 + threadIdx.x;   // 16,777,216 quads
  int n0 = (int)(q & 31) * 4;
  int m  = (int)((q >> 5) & 127);
  size_t img = q >> 12;                                // b*512+ch
  const float* yb = ypad + img * (YP * YP);
  float a0 = 0.f, a1 = 0.f, a2 = 0.f, a3 = 0.f;
  const float tw[4] = {1.f, 3.f, 3.f, 1.f};
  #pragma unroll
  for (int fy = 0; fy < 4; ++fy) {
    const float* r = yb + (size_t)(m + fy) * YP + n0;
    float r0 = r[0], r1 = r[1], r2 = r[2], r3 = r[3], r4 = r[4], r5 = r[5], r6 = r[6];
    float wgt = tw[fy];
    a0 += wgt * (r0 + 3.f * r1 + 3.f * r2 + r3);
    a1 += wgt * (r1 + 3.f * r2 + 3.f * r3 + r4);
    a2 += wgt * (r2 + 3.f * r3 + 3.f * r4 + r5);
    a3 += wgt * (r3 + 3.f * r4 + 3.f * r5 + r6);
  }
  float* o = out + img * (128 * 128) + (size_t)m * 128 + n0;
  o[0] = a0 * 0.0625f;
  o[1] = a1 * 0.0625f;
  o[2] = a2 * 0.0625f;
  o[3] = a3 * 0.0625f;
}

// ---------------------------------------------------------------- launch
extern "C" void kernel_launch(void* const* d_in, const int* in_sizes, int n_in,
                              void* d_out, int out_size, void* d_ws, size_t ws_size,
                              hipStream_t stream) {
  const float* x     = (const float*)d_in[0];
  const float* style = (const float*)d_in[1];
  const float* w     = (const float*)d_in[2];
  const float* mod_w = (const float*)d_in[3];
  const float* mod_b = (const float*)d_in[4];
  float* out = (float*)d_out;

  char* ws = (char*)d_ws;
  float*          s_buf = (float*)(ws + WS_S_OFF);
  unsigned short* A_buf = (unsigned short*)(ws + WS_A_OFF);
  float*          y_buf = (float*)(ws + WS_Y_OFF);

  style_kernel<<<16, 256, 0, stream>>>(style, mod_w, mod_b, s_buf);
  modw_kernel<<<B_ * FO_, 256, 0, stream>>>(w, s_buf, A_buf);
  zero_border_kernel<<<B_ * FO_, 256, 0, stream>>>(y_buf);
  dim3 g3(17 * 5, FO_ / 32, B_);
  conv_kernel<<<g3, 256, 0, stream>>>(x, A_buf, y_buf);
  fir_kernel<<<65536, 256, 0, stream>>>(y_buf, out);
}
